// SignDenoising_72146860638719
// MI455X (gfx1250) — matrix operand order repaired and checked by reading, not compiled, not run on hardware
//
#include <hip/hip_runtime.h>

typedef __attribute__((ext_vector_type(16))) _Float16 v16h;
typedef __attribute__((ext_vector_type(8)))  float    v8f;

// ---------------------------------------------------------------- utilities

__global__ void zero_f32(float* p, long long n) {
    long long i = (long long)blockIdx.x * blockDim.x + threadIdx.x;
    if (i < n) p[i] = 0.0f;
}

// p[i*width + c] = bias[c]; width must be a power of two (64 / 2 here)
__global__ void init_rows(float* p, const float* bias, long long tot, int wmask) {
    long long i = (long long)blockIdx.x * blockDim.x + threadIdx.x;
    if (i < tot) p[i] = bias[(int)(i & wmask)];
}

// ------------------------------------------------- column mean of x [N,64]

__global__ void colsum_kernel(const float* __restrict__ x, float* __restrict__ mean, int n) {
    __shared__ float part[64];
    int tid = threadIdx.x;                 // 256 threads
    if (tid < 64) part[tid] = 0.0f;
    __syncthreads();
    int col  = tid & 63;
    int rgrp = tid >> 6;                   // 4 row groups per block
    float acc = 0.0f;
    for (int r = blockIdx.x * 4 + rgrp; r < n; r += gridDim.x * 4)
        acc += x[(size_t)r * 64 + col];
    atomicAdd(&part[col], acc);
    __syncthreads();
    if (tid < 64) atomicAdd(&mean[tid], part[tid]);
}

__global__ void mean_div_kernel(float* mean, float inv_n) {
    int t = threadIdx.x;
    if (t < 64) mean[t] *= inv_n;
}

// c[j] = sum_k mean[k] * W1[(64+k)*64 + j]   (the "concat mean" half of GEMM1)
__global__ void cvec_kernel(const float* __restrict__ mean, const float* __restrict__ W1,
                            float* __restrict__ cvec) {
    int j = threadIdx.x;
    if (j < 64) {
        float a = 0.0f;
        for (int k = 0; k < 64; ++k) a += mean[k] * W1[(64 + k) * 64 + j];
        cvec[j] = a;
    }
}

// -------------------------------------------------------- degree / dis[n]

__global__ void degree_kernel(const int* __restrict__ dst, float* __restrict__ deg, long long E) {
    long long e = (long long)blockIdx.x * blockDim.x + threadIdx.x;
    if (e < E) atomicAdd(&deg[dst[e]], 1.0f);
}

__global__ void dis_kernel(float* dis, int n) {   // in-place deg -> dis
    int i = blockIdx.x * blockDim.x + threadIdx.x;
    if (i < n) {
        float d = dis[i];
        dis[i] = (d > 0.0f) ? rsqrtf(d) : 0.0f;
    }
}

// ------------------------------------- GEMM1 (WMMA): h1 = x @ W1a + cvec
// One wave -> 16 rows x 64 cols.  4 waves / block (128 threads).

__global__ void __launch_bounds__(128)
gemm1_wmma_kernel(const float* __restrict__ x, const float* __restrict__ W1,
                  const float* __restrict__ cvec, float* __restrict__ h1, int n) {
    __shared__ _Float16 Bh[64 * 64];       // W1a as f16, [k][n], 8 KB
    __shared__ _Float16 Ah[4][16 * 64];    // per-wave 16x64 x-tile f16, 8 KB

    const int tid  = threadIdx.x;
    const int wave = tid >> 5;
    const int lane = tid & 31;

    // stage W1a (in-features 0..63) -> LDS f16
    for (int i = tid; i < 64 * 64; i += 128) Bh[i] = (_Float16)W1[i];

    const int row0 = (blockIdx.x * 4 + wave) * 16;

    // stage this wave's 16x64 tile of x -> LDS f16 (zero-pad past n)
    for (int i = lane; i < 16 * 64; i += 32) {
        int m = i >> 6, k = i & 63;
        int r = row0 + m;
        Ah[wave][i] = (_Float16)((r < n) ? x[(size_t)r * 64 + k] : 0.0f);
    }
    __syncthreads();

    const int mrow = lane & 15;            // A-matrix row (M) for this lane
    const int khi  = lane >> 4;            // lane group selects K-half
    const int ncol = lane & 15;            // B/C/D column (N)
    const bool full_tile = (row0 + 16 <= n);

    // A fragments: 16-bit A layout -> half i=2v+p holds K = base16(v) + (v&3)*2 + p + 8*khi
    v16h a0, a1;
    #pragma unroll
    for (int i = 0; i < 16; ++i) {
        int v = i >> 1, p = i & 1;
        int k = ((v < 4) ? 0 : 16) + (v & 3) * 2 + p + khi * 8;
        a0[i] = Ah[wave][mrow * 64 + k];        // K =  0..31
        a1[i] = Ah[wave][mrow * 64 + 32 + k];   // K = 32..63
    }

    // row-base pointer for D stores: lane's column within the tile row
    float* const dbase = h1 + (size_t)(row0 + khi * 8) * 64 + ncol;

    #pragma unroll
    for (int nt = 0; nt < 4; ++nt) {       // four 16-wide N tiles -> 64 outputs
        // B fragments: 16-bit B layout -> half i holds K = (i<8 ? i : i+8) + 8*khi, N = ncol
        v16h b0, b1;
        #pragma unroll
        for (int i = 0; i < 16; ++i) {
            int k = ((i < 8) ? i : i + 8) + khi * 8;
            b0[i] = Bh[k * 64        + nt * 16 + ncol];
            b1[i] = Bh[(k + 32) * 64 + nt * 16 + ncol];
        }
        // accumulator preloaded with broadcast constant c[n] (same for all M)
        v8f acc;
        float cv = cvec[nt * 16 + ncol];
        #pragma unroll
        for (int r = 0; r < 8; ++r) acc[r] = cv;

        acc = __builtin_amdgcn_wmma_f32_16x16x32_f16(false, a0, false, b0,
                                                     (short)0, acc, false, false);
        acc = __builtin_amdgcn_wmma_f32_16x16x32_f16(false, a1, false, b1,
                                                     (short)0, acc, false, false);

        // C/D layout: element r -> M = r + 8*khi, N = ncol
        if (full_tile) {                   // wave-uniform: unguarded stores
            #pragma unroll
            for (int r = 0; r < 8; ++r) dbase[r * 64 + nt * 16] = acc[r];
        } else {                           // boundary tile only
            for (int r = 0; r < 8; ++r) {
                int row = row0 + r + khi * 8;
                if (row < n) h1[(size_t)row * 64 + nt * 16 + ncol] = acc[r];
            }
        }
    }
}

// ------------------- edge scatter, layer 1: agg[dst] += h1[src] * norm
// 16 threads per edge, float4 gather, 4 f32 atomics (all L2-resident).

__global__ void scatter1_kernel(const int* __restrict__ src, const int* __restrict__ dst,
                                const float* __restrict__ dis, const float* __restrict__ h1,
                                float* __restrict__ agg, long long E) {
    long long gt = (long long)blockIdx.x * blockDim.x + threadIdx.x;
    if (gt >= E * 16) return;
    long long e = gt >> 4;
    int seg = (int)(gt & 15);
    int s = src[e], d = dst[e];
    float nrm = dis[s] * dis[d];
    if (nrm != 0.0f) {
        const float4 v = *(const float4*)(h1 + (size_t)s * 64 + seg * 4);
        float* o = agg + (size_t)d * 64 + seg * 4;
        atomicAdd(o + 0, v.x * nrm);
        atomicAdd(o + 1, v.y * nrm);
        atomicAdd(o + 2, v.z * nrm);
        atomicAdd(o + 3, v.w * nrm);
    }
}

// ------------- GEMM2 with fused ReLU: h2 = relu(agg1) @ W2   ([64,2], tiny)

__global__ void gemm2_relu_kernel(const float* __restrict__ agg1, const float* __restrict__ W2,
                                  float* __restrict__ h2, int n) {
    __shared__ float w[128];
    if (threadIdx.x < 128) w[threadIdx.x] = W2[threadIdx.x];
    __syncthreads();
    int i = blockIdx.x * blockDim.x + threadIdx.x;
    if (i >= n) return;
    const float4* r4 = (const float4*)(agg1 + (size_t)i * 64);
    float a0 = 0.0f, a1 = 0.0f;
    #pragma unroll
    for (int q = 0; q < 16; ++q) {         // 16 x float4 = 64 features
        float4 v = r4[q];
        float v0 = fmaxf(v.x, 0.0f), v1 = fmaxf(v.y, 0.0f);
        float v2 = fmaxf(v.z, 0.0f), v3 = fmaxf(v.w, 0.0f);
        int k = q * 4;
        a0 += v0 * w[(k + 0) * 2] + v1 * w[(k + 1) * 2] + v2 * w[(k + 2) * 2] + v3 * w[(k + 3) * 2];
        a1 += v0 * w[(k + 0) * 2 + 1] + v1 * w[(k + 1) * 2 + 1]
            + v2 * w[(k + 2) * 2 + 1] + v3 * w[(k + 3) * 2 + 1];
    }
    h2[(size_t)i * 2]     = a0;
    h2[(size_t)i * 2 + 1] = a1;
}

// ----------------- edge scatter, layer 2: out[dst] += h2[src] * norm

__global__ void scatter2_kernel(const int* __restrict__ src, const int* __restrict__ dst,
                                const float* __restrict__ dis, const float* __restrict__ h2,
                                float* __restrict__ out, long long E) {
    long long e = (long long)blockIdx.x * blockDim.x + threadIdx.x;
    if (e >= E) return;
    int s = src[e], d = dst[e];
    float nrm = dis[s] * dis[d];
    if (nrm != 0.0f) {
        atomicAdd(&out[(size_t)d * 2],     h2[(size_t)s * 2]     * nrm);
        atomicAdd(&out[(size_t)d * 2 + 1], h2[(size_t)s * 2 + 1] * nrm);
    }
}

// ---------------------------------------------------------------- launcher

extern "C" void kernel_launch(void* const* d_in, const int* in_sizes, int n_in,
                              void* d_out, int out_size, void* d_ws, size_t ws_size,
                              hipStream_t stream) {
    const float* x  = (const float*)d_in[0];
    const int*   ei = (const int*)d_in[1];        // int32 (JAX x64 disabled)
    const float* W1 = (const float*)d_in[2];
    const float* b1 = (const float*)d_in[3];
    const float* W2 = (const float*)d_in[4];
    const float* b2 = (const float*)d_in[5];

    const int       N = in_sizes[0] / 64;
    const long long E = (long long)in_sizes[1] / 2;
    const int* src = ei;
    const int* dst = ei + E;

    // workspace layout (floats)
    char* ws = (char*)d_ws;
    float* dis  = (float*)ws;                                   ws += ((size_t)N * 4 + 255) & ~255ull;
    float* mean = (float*)ws;                                   ws += 256;
    float* cvec = (float*)ws;                                   ws += 256;
    float* h1   = (float*)ws;                                   ws += (size_t)N * 64 * 4;
    float* agg1 = (float*)ws;                                   ws += (size_t)N * 64 * 4;
    float* h2   = (float*)ws;
    float* out  = (float*)d_out;

    const int B = 256;

    // 1) zero deg + mean accumulators
    zero_f32<<<(N + B - 1) / B, B, 0, stream>>>(dis, N);
    zero_f32<<<1, 64, 0, stream>>>(mean, 64);

    // 2) column mean of x, then cvec = mean @ W1[64:128]
    colsum_kernel<<<256, 256, 0, stream>>>(x, mean, N);
    mean_div_kernel<<<1, 64, 0, stream>>>(mean, 1.0f / (float)N);
    cvec_kernel<<<1, 64, 0, stream>>>(mean, W1, cvec);

    // 3) degree -> dis (symmetric GCN normalization)
    degree_kernel<<<(int)((E + B - 1) / B), B, 0, stream>>>(dst, dis, E);
    dis_kernel<<<(N + B - 1) / B, B, 0, stream>>>(dis, N);

    // 4) h1 = x @ W1a + cvec   (WMMA f16->f32; 16 rows per wave, 4 waves/block)
    gemm1_wmma_kernel<<<(N + 63) / 64, 128, 0, stream>>>(x, W1, cvec, h1, N);

    // 5) agg1 = b1 (broadcast), then scatter-add edge messages
    init_rows<<<(int)(((long long)N * 64 + B - 1) / B), B, 0, stream>>>(agg1, b1, (long long)N * 64, 63);
    scatter1_kernel<<<(int)((E * 16 + B - 1) / B), B, 0, stream>>>(src, dst, dis, h1, agg1, E);

    // 6) h2 = relu(agg1) @ W2
    gemm2_relu_kernel<<<(N + B - 1) / B, B, 0, stream>>>(agg1, W2, h2, N);

    // 7) out = b2 (broadcast), then scatter-add layer-2 messages
    init_rows<<<(int)(((long long)N * 2 + B - 1) / B), B, 0, stream>>>(out, b2, (long long)N * 2, 1);
    scatter2_kernel<<<(int)((E + B - 1) / B), B, 0, stream>>>(src, dst, dis, h2, out, E);
}
